// _FP8LinearBase_33243046871359
// MI455X (gfx1250) — 8-bit matrix operand order repaired and checked by reading, compiled, not run on hardware
//
#include <hip/hip_runtime.h>
#include <stdint.h>


typedef __attribute__((ext_vector_type(16))) int   v16i;
typedef __attribute__((ext_vector_type(4)))  int   v4i;
typedef __attribute__((ext_vector_type(8)))  float v8f;

#define M_DIM 4096
#define N_DIM 4096
#define K_DIM 4096
#define KB    32
#define LDSS  144
#define TILE_BYTES (128 * LDSS)

__device__ __forceinline__ float bf16_bits_to_f32(unsigned int hi16_in_low) {
    return __uint_as_float(hi16_in_low << 16);
}

__device__ __forceinline__ unsigned short f32_to_bf16_rne(float f) {
    unsigned int u = __float_as_uint(f);
    u = (u + 0x7fffu + ((u >> 16) & 1u)) >> 16;
    return (unsigned short)u;
}

#if defined(__has_builtin)
#if __has_builtin(__builtin_amdgcn_cvt_pk_fp8_f32)
#define HAVE_CVT_PK_FP8 1
#endif
#if __has_builtin(__builtin_amdgcn_global_load_async_to_lds_b128)
#define HAVE_ASYNC_LDS 1
#endif
#endif

#ifdef HAVE_ASYNC_LDS
#define AS1P(p) ((__attribute__((address_space(1))) v4i*)(p))
#define AS3P(p) ((__attribute__((address_space(3))) v4i*)(p))
#if __has_builtin(__builtin_amdgcn_s_wait_asynccnt)
#define WAIT_ASYNC(n) __builtin_amdgcn_s_wait_asynccnt(n)
#else
#define WAIT_ASYNC(n) asm volatile("s_wait_asynccnt " #n ::: "memory")
#endif
#else
#define WAIT_ASYNC(n)
#endif

__device__ __forceinline__ unsigned int f32_to_e4m3_sw(float f) {
    unsigned int u = __float_as_uint(f);
    unsigned int s = (u >> 24) & 0x80u;
    float a = fabsf(f);
    if (!(a == a)) return s | 0x7Fu;
    if (a >= 448.0f) return s | 0x7Eu;
    if (a < 0.015625f) {
        int m = (int)rintf(a * 512.0f);
        return s | (unsigned int)m;
    }
    unsigned int bits = __float_as_uint(a);
    unsigned int exp  = bits >> 23;
    unsigned int mant = bits & 0x7fffffu;
    unsigned int q    = mant >> 20;
    unsigned int rem  = mant & 0xfffffu;
    if (rem > 0x80000u || (rem == 0x80000u && (q & 1u))) q++;
    unsigned int E = exp - 127u + 7u;
    if (q == 8u) { q = 0u; E++; }
    unsigned int code = (E << 3) | q;
    if (code > 0x7Eu) code = 0x7Eu;
    return s | code;
}

__device__ __forceinline__ unsigned int pack_fp8x4(float a, float b, float c, float d) {
#ifdef HAVE_CVT_PK_FP8
    int p = 0;
    p = __builtin_amdgcn_cvt_pk_fp8_f32(a, b, p, false);
    p = __builtin_amdgcn_cvt_pk_fp8_f32(c, d, p, true);
    return (unsigned int)p;
#else
    return f32_to_e4m3_sw(a) | (f32_to_e4m3_sw(b) << 8) |
           (f32_to_e4m3_sw(c) << 16) | (f32_to_e4m3_sw(d) << 24);
#endif
}

__global__ __launch_bounds__(256) void quant_act_kernel(
    const unsigned short* __restrict__ x,
    unsigned char* __restrict__ xq,
    float* __restrict__ xs)
{
    const int lane  = threadIdx.x & 31;
    const int gwave = (blockIdx.x * blockDim.x + threadIdx.x) >> 5;
    const int m  = gwave >> 5;
    const int kb = gwave & 31;

    const unsigned short* p = x + (size_t)m * K_DIM + kb * 128 + lane * 4;
    const uint2 v = *reinterpret_cast<const uint2*>(p);

    const float f0 = bf16_bits_to_f32(v.x & 0xffffu);
    const float f1 = __uint_as_float(v.x & 0xffff0000u);
    const float f2 = bf16_bits_to_f32(v.y & 0xffffu);
    const float f3 = __uint_as_float(v.y & 0xffff0000u);

    float amax = fmaxf(fmaxf(fabsf(f0), fabsf(f1)), fmaxf(fabsf(f2), fabsf(f3)));
#pragma unroll
    for (int off = 16; off >= 1; off >>= 1)
        amax = fmaxf(amax, __shfl_xor(amax, off, 32));

    const float safe = fmaxf(amax, 1e-4f);
    const float r    = 448.0f / safe;

    const unsigned int pk = pack_fp8x4(f0 * r, f1 * r, f2 * r, f3 * r);
    reinterpret_cast<unsigned int*>(xq)[(size_t)m * (K_DIM / 4) + kb * 32 + lane] = pk;
    if (lane == 0) xs[m * KB + kb] = safe * (1.0f / 448.0f);
}

__global__ __launch_bounds__(256) void quant_wgt_kernel(
    const unsigned short* __restrict__ w,
    unsigned char* __restrict__ wq,
    float* __restrict__ ws)
{
    __shared__ float red[256];
    const int t  = threadIdx.x;
    const int nb = blockIdx.y;
    const int kb = blockIdx.x;

    const int row  = nb * 128 + (t >> 1);
    const int col0 = kb * 128 + (t & 1) * 64;
    const unsigned short* p = w + (size_t)row * K_DIM + col0;

    uint4 buf[8];
    float amax = 0.0f;
#pragma unroll
    for (int i = 0; i < 8; ++i) {
        buf[i] = *reinterpret_cast<const uint4*>(p + i * 8);
        const unsigned int dw[4] = {buf[i].x, buf[i].y, buf[i].z, buf[i].w};
#pragma unroll
        for (int j = 0; j < 4; ++j) {
            amax = fmaxf(amax, fabsf(bf16_bits_to_f32(dw[j] & 0xffffu)));
            amax = fmaxf(amax, fabsf(__uint_as_float(dw[j] & 0xffff0000u)));
        }
    }
    red[t] = amax;
    __syncthreads();
#pragma unroll
    for (int s = 128; s > 0; s >>= 1) {
        if (t < s) red[t] = fmaxf(red[t], red[t + s]);
        __syncthreads();
    }
    const float safe = fmaxf(red[0], 1e-6f);
    const float r    = 448.0f / safe;
    if (t == 0) ws[nb * KB + kb] = safe * (1.0f / 448.0f);

    unsigned char* q = wq + (size_t)row * K_DIM + col0;
#pragma unroll
    for (int i = 0; i < 8; ++i) {
        const unsigned int dw[4] = {buf[i].x, buf[i].y, buf[i].z, buf[i].w};
        float f[8];
#pragma unroll
        for (int j = 0; j < 4; ++j) {
            f[2 * j]     = bf16_bits_to_f32(dw[j] & 0xffffu) * r;
            f[2 * j + 1] = __uint_as_float(dw[j] & 0xffff0000u) * r;
        }
        uint2 o;
        o.x = pack_fp8x4(f[0], f[1], f[2], f[3]);
        o.y = pack_fp8x4(f[4], f[5], f[6], f[7]);
        *reinterpret_cast<uint2*>(q + i * 8) = o;
    }
}


__device__ __forceinline__ void issue_stage(
    const unsigned char* __restrict__ xq,
    const unsigned char* __restrict__ wq,
    unsigned char* sA, unsigned char* sB,
    int t, int tile_m, int tile_n, int kb)
{
#pragma unroll
    for (int i = 0; i < 4; ++i) {
        const int c   = t + i * 256;
        const int row = c >> 3;
        const int col = (c & 7) << 4;
        const unsigned char* ga = &xq[(size_t)(tile_m + row) * K_DIM + kb * 128 + col];
        const unsigned char* gb = &wq[(size_t)(tile_n + row) * K_DIM + kb * 128 + col];
        unsigned char* la = &sA[row * LDSS + col];
        unsigned char* lb = &sB[row * LDSS + col];
#ifdef HAVE_ASYNC_LDS
        __builtin_amdgcn_global_load_async_to_lds_b128(AS1P(ga), AS3P(la), 0, 0);
        __builtin_amdgcn_global_load_async_to_lds_b128(AS1P(gb), AS3P(lb), 0, 0);
#else
        *reinterpret_cast<uint4*>(la) = *reinterpret_cast<const uint4*>(ga);
        *reinterpret_cast<uint4*>(lb) = *reinterpret_cast<const uint4*>(gb);
#endif
    }
}

__global__ __launch_bounds__(256) void fp8_gemm_kernel(
    const unsigned char* __restrict__ xq,
    const unsigned char* __restrict__ wq,
    const float* __restrict__ xs,
    const float* __restrict__ ws,
    unsigned short* __restrict__ out)
{
    __shared__ alignas(16) unsigned char sA[2 * TILE_BYTES];
    __shared__ alignas(16) unsigned char sB[2 * TILE_BYTES];
    __shared__ float sXS[128 * KB];

    const int t    = threadIdx.x;
    const int lane = t & 31;
    const int wave = t >> 5;
    const int wm   = wave & 3;
    const int wn   = wave >> 2;
    const int lr   = lane & 15;
    const int hi   = lane >> 4;

    const int tile_m = blockIdx.y * 128;
    const int tile_n = blockIdx.x * 128;

    issue_stage(xq, wq, sA, sB, t, tile_m, tile_n, 0);

#pragma unroll
    for (int i = 0; i < 16; ++i) {
        const int idx = t + i * 256;
        sXS[idx] = xs[(size_t)tile_m * KB + idx];
    }

    v8f acc[2][4];
#pragma unroll
    for (int i = 0; i < 2; ++i)
#pragma unroll
        for (int j = 0; j < 4; ++j)
#pragma unroll
            for (int g = 0; g < 8; ++g) acc[i][j][g] = 0.0f;

    const v8f czero = {};

    for (int kb = 0; kb < KB; ++kb) {
        const int cur = kb & 1;
        if (kb + 1 < KB) {
            issue_stage(xq, wq, sA + (1 - cur) * TILE_BYTES, sB + (1 - cur) * TILE_BYTES,
                        t, tile_m, tile_n, kb + 1);
            WAIT_ASYNC(8);
        } else {
            WAIT_ASYNC(0);
        }
        __syncthreads();

        const unsigned char* cA = sA + cur * TILE_BYTES;
        const unsigned char* cB = sB + cur * TILE_BYTES;

        const float wscale = ws[(tile_n >> 7) * KB + kb];

        v16i a[2];
#pragma unroll
        for (int i = 0; i < 2; ++i) {
            const unsigned char* base = &cA[(wm * 32 + i * 16 + lr) * LDSS + hi * 16];
#pragma unroll
            for (int jj = 0; jj < 4; ++jj) {
                const int4 v = *reinterpret_cast<const int4*>(base + jj * 32);
                a[i][4 * jj]     = v.x;
                a[i][4 * jj + 1] = v.y;
                a[i][4 * jj + 2] = v.z;
                a[i][4 * jj + 3] = v.w;
            }
        }

        v16i b[4];
#pragma unroll
        for (int j = 0; j < 4; ++j) {
            const unsigned char* base = &cB[(wn * 64 + j * 16 + lr) * LDSS + hi * 16];
#pragma unroll
            for (int g = 0; g < 4; ++g) {
                const int4 v = *reinterpret_cast<const int4*>(base + g * 32);
                b[j][4 * g]     = v.x;
                b[j][4 * g + 1] = v.y;
                b[j][4 * g + 2] = v.z;
                b[j][4 * g + 3] = v.w;
            }
        }

        float sw[2][8];
#pragma unroll
        for (int i = 0; i < 2; ++i)
#pragma unroll
            for (int g = 0; g < 8; ++g)
                sw[i][g] = sXS[(wm * 32 + i * 16 + g + hi * 8) * KB + kb] * wscale;

#pragma unroll
        for (int i = 0; i < 2; ++i) {
#pragma unroll
            for (int j = 0; j < 4; ++j) {
                const v8f d = __builtin_amdgcn_wmma_f32_16x16x128_fp8_fp8(
                    a[i], b[j], (short)0, czero, false, false);
#pragma unroll
                for (int g = 0; g < 8; ++g)
                    acc[i][j][g] += sw[i][g] * d[g];
            }
        }
        __syncthreads();
    }

#pragma unroll
    for (int i = 0; i < 2; ++i)
#pragma unroll
        for (int j = 0; j < 4; ++j)
#pragma unroll
            for (int g = 0; g < 8; ++g) {
                const int row = tile_m + wm * 32 + i * 16 + g + hi * 8;
                const int col = tile_n + wn * 64 + j * 16 + lr;
                out[(size_t)row * N_DIM + col] = f32_to_bf16_rne(acc[i][j][g]);
            }
}

extern "C" void kernel_launch(void* const* d_in, const int* in_sizes, int n_in,
                              void* d_out, int out_size, void* d_ws, size_t ws_size,
                              hipStream_t stream) {
    (void)in_sizes; (void)n_in; (void)out_size; (void)ws_size;

    const unsigned short* x = (const unsigned short*)d_in[0];
    const unsigned short* w = (const unsigned short*)d_in[1];
    unsigned short* out     = (unsigned short*)d_out;

    unsigned char* xq = (unsigned char*)d_ws;
    unsigned char* wq = xq + (size_t)M_DIM * K_DIM;
    float* xs         = (float*)(wq + (size_t)N_DIM * K_DIM);
    float* ws         = xs + (size_t)M_DIM * KB;

    quant_act_kernel<<<dim3((M_DIM * KB) / 8), dim3(256), 0, stream>>>(x, xq, xs);

    quant_wgt_kernel<<<dim3(KB, N_DIM / 128), dim3(256), 0, stream>>>(w, wq, ws);

    fp8_gemm_kernel<<<dim3(N_DIM / 128, M_DIM / 128), dim3(256), 0, stream>>>(
        xq, wq, xs, ws, out);
}
